// SpeechLMTransformerBlock_26456998543936
// MI455X (gfx1250) — matrix operand order repaired and checked by reading, not compiled, not run on hardware
//
#include <hip/hip_runtime.h>
#include <hip/hip_bf16.h>

// ---------------------------------------------------------------------------
// SpeechLM transformer block for MI455X (gfx1250, wave32, WMMA).
// - All GEMMs + both attention contractions on v_wmma_f32_16x16x32_bf16
// - Double-buffered global->LDS staging via global_load_async_to_lds_b128
//   (ASYNCcnt in-order completion => s_wait_asynccnt 8 drains previous tile)
// - Weights pre-transposed to [N][K] bf16 so every LDS tile is a plain 2D copy
// ---------------------------------------------------------------------------

#define Hdim  1024
#define NHEAD 16
#define HDH   64      // head dim
#define Bb    2
#define Ss    2048
#define Cc    256

typedef __attribute__((ext_vector_type(16))) __bf16 v16bf;
typedef __attribute__((ext_vector_type(8)))  __bf16 v8bf;
typedef __attribute__((ext_vector_type(8)))  float  v8f;
typedef __attribute__((ext_vector_type(4)))  unsigned int v4u;
typedef __attribute__((ext_vector_type(8)))  int v8i;
typedef __attribute__((ext_vector_type(4)))  int v4i;

__device__ __forceinline__ unsigned short f32_to_bf16(float f) {
    unsigned int u = __float_as_uint(f);
    u = (u + 0x7FFFu + ((u >> 16) & 1u)) >> 16;
    return (unsigned short)u;
}

// Build a 16-element bf16 A/B fragment from two 16-byte aligned LDS chunks.
__device__ __forceinline__ v16bf make_frag(const unsigned short* lo,
                                           const unsigned short* hi) {
    v8bf a = *(const v8bf*)lo;
    v8bf b = *(const v8bf*)hi;
    v16bf r;
#pragma unroll
    for (int i = 0; i < 8; ++i) { r[i] = a[i]; r[i + 8] = b[i]; }
    return r;
}

// Async 16B global -> LDS copy (per lane). VDST = 32-bit LDS byte offset
// (low 32 bits of the flat shared address), VADDR = 64-bit global address.
__device__ __forceinline__ void async_copy16(const unsigned short* lds,
                                             const unsigned short* g) {
    unsigned loff = (unsigned)(unsigned long long)lds;   // LDS aperture: low 32 = offset
    unsigned long long ga = (unsigned long long)g;
    asm volatile("global_load_async_to_lds_b128 %0, %1, off"
                 :: "v"(loff), "v"(ga) : "memory");
}
__device__ __forceinline__ void async_wait0() {
    asm volatile("s_wait_asynccnt 0x0" ::: "memory");
}
__device__ __forceinline__ void async_wait4() {
    asm volatile("s_wait_asynccnt 0x4" ::: "memory");
}
__device__ __forceinline__ void async_wait8() {
    asm volatile("s_wait_asynccnt 0x8" ::: "memory");
}

// ---------------------------------------------------------------------------
// f32 -> bf16 conversion (conditioning)
// ---------------------------------------------------------------------------
__global__ void convert_bf16_kernel(const float* __restrict__ in,
                                    unsigned short* __restrict__ out,
                                    size_t n) {
    size_t i = (size_t)blockIdx.x * blockDim.x + threadIdx.x;
    size_t st = (size_t)gridDim.x * blockDim.x;
    for (; i < n; i += st) out[i] = f32_to_bf16(in[i]);
}

// ---------------------------------------------------------------------------
// Weight transpose + convert: W[K,N] f32 -> Wt[N,K] bf16 (32x32 LDS tiles)
// ---------------------------------------------------------------------------
__global__ __launch_bounds__(256)
void transpose_convert_kernel(const float* __restrict__ in,
                              unsigned short* __restrict__ out,
                              int K, int N) {
    __shared__ float t[32][33];
    const int nb = blockIdx.x * 32, kb = blockIdx.y * 32;
    const int tx = threadIdx.x & 31, ty = threadIdx.x >> 5;
#pragma unroll
    for (int r = ty; r < 32; r += 8)
        t[r][tx] = in[(size_t)(kb + r) * N + nb + tx];
    __syncthreads();
#pragma unroll
    for (int r = ty; r < 32; r += 8)
        out[(size_t)(nb + r) * K + kb + tx] = f32_to_bf16(t[tx][r]);
}

// ---------------------------------------------------------------------------
// LayerNorm: one row (H=1024) per 256-thread block, bf16 output
// ---------------------------------------------------------------------------
__global__ __launch_bounds__(256)
void layernorm_kernel(const float* __restrict__ X,
                      const float* __restrict__ g,
                      const float* __restrict__ bt,
                      unsigned short* __restrict__ Out) {
    __shared__ float red[16];
    const int row = blockIdx.x;
    const float* xr = X + (size_t)row * Hdim;
    float s = 0.f, ss = 0.f;
    for (int i = threadIdx.x; i < Hdim; i += 256) {
        float v = xr[i]; s += v; ss += v * v;
    }
#pragma unroll
    for (int off = 16; off; off >>= 1) {
        s  += __shfl_xor(s,  off, 32);
        ss += __shfl_xor(ss, off, 32);
    }
    const int wave = threadIdx.x >> 5, lane = threadIdx.x & 31;
    if (lane == 0) { red[wave] = s; red[8 + wave] = ss; }
    __syncthreads();
    if (threadIdx.x == 0) {
        float ts = 0.f, tss = 0.f;
        for (int w = 0; w < 8; ++w) { ts += red[w]; tss += red[8 + w]; }
        red[0] = ts; red[8] = tss;
    }
    __syncthreads();
    const float mean = red[0] * (1.0f / Hdim);
    const float var  = red[8] * (1.0f / Hdim) - mean * mean;
    const float inv  = rsqrtf(var + 1e-5f);
    for (int i = threadIdx.x; i < Hdim; i += 256) {
        float v = (xr[i] - mean) * inv * g[i] + bt[i];
        Out[(size_t)row * Hdim + i] = f32_to_bf16(v);
    }
}

// ---------------------------------------------------------------------------
// RoPE + head split: qkv f32 [B,S,3H] -> Q,K (roped) [B,NH,S,HD],
//                    V -> d-major [B,NH,HD,S]   (all bf16)
// ---------------------------------------------------------------------------
__global__ __launch_bounds__(256)
void rope_split_kernel(const float* __restrict__ qkv,
                       const float* __restrict__ freqs,
                       unsigned short* __restrict__ Qo,
                       unsigned short* __restrict__ Ko,
                       unsigned short* __restrict__ Vt) {
    const int s = blockIdx.x, b = blockIdx.y;
    const float* base = qkv + ((size_t)b * Ss + s) * (3 * Hdim);
    for (int p = threadIdx.x; p < NHEAD * (HDH / 2); p += 256) {
        const int h = p >> 5, d = p & 31;
        const float fr = freqs[s * (HDH / 2) + d];
        const float cs = cosf(fr), sn = sinf(fr);
        const size_t o = ((size_t)(b * NHEAD + h) * Ss + s) * HDH + d;
        float q1 = base[h * HDH + d], q2 = base[h * HDH + d + 32];
        Qo[o]      = f32_to_bf16(q1 * cs - q2 * sn);
        Qo[o + 32] = f32_to_bf16(q2 * cs + q1 * sn);
        float k1 = base[Hdim + h * HDH + d], k2 = base[Hdim + h * HDH + d + 32];
        Ko[o]      = f32_to_bf16(k1 * cs - k2 * sn);
        Ko[o + 32] = f32_to_bf16(k2 * cs + k1 * sn);
    }
    for (int e = threadIdx.x; e < Hdim; e += 256) {
        const int h = e >> 6, d = e & 63;
        Vt[((size_t)(b * NHEAD + h) * HDH + d) * Ss + s] =
            f32_to_bf16(base[2 * Hdim + e]);
    }
}

// ---------------------------------------------------------------------------
// f32 [B,L,nmat*H] (pick matrix mi) -> bf16 heads.
// transpose==0: [B,NH,L,HD]; transpose==1: [B,NH,HD,L] (d-major, for V)
// ---------------------------------------------------------------------------
__global__ __launch_bounds__(256)
void reshape_heads_kernel(const float* __restrict__ in,
                          unsigned short* __restrict__ out,
                          int L, int nmat, int mi, int transpose) {
    const int s = blockIdx.x, b = blockIdx.y;
    const float* base = in + ((size_t)b * L + s) * (size_t)(nmat * Hdim) + (size_t)mi * Hdim;
    for (int e = threadIdx.x; e < Hdim; e += 256) {
        const int h = e >> 6, d = e & 63;
        const size_t o = transpose
            ? ((size_t)(b * NHEAD + h) * HDH + d) * L + s
            : ((size_t)(b * NHEAD + h) * L + s) * HDH + d;
        out[o] = f32_to_bf16(base[e]);
    }
}

// ---------------------------------------------------------------------------
// Tiled WMMA GEMM: C[M,N] = A[M,K](bf16) * Wt[N,K](bf16, pre-transposed)
// Block tile 128x128, BK=64, 8 waves (2M x 4N), wave tile 64x32.
// Double-buffered async DMA staging; templated epilogue (no runtime branches).
// ---------------------------------------------------------------------------
#define GLDS 72   // padded row stride (bf16 elems)

template<int BIAS, int RES, int GELU, int OUTF, int OUTB>
__global__ __launch_bounds__(256)
void gemm_bf16_kernel(const unsigned short* __restrict__ A,
                      const unsigned short* __restrict__ Wt,
                      const float* __restrict__ bias,
                      const float* __restrict__ residual,
                      float* __restrict__ Cf,
                      unsigned short* __restrict__ Cbf,
                      int M, int N, int K) {
    __shared__ __align__(16) unsigned short sA[2][128 * GLDS];
    __shared__ __align__(16) unsigned short sB[2][128 * GLDS];

    const int tid = threadIdx.x;
    const int wave = tid >> 5, lane = tid & 31;
    const int lh = lane >> 4, ln = lane & 15;
    const int wm = wave & 1, wn = wave >> 1;           // 2 x 4 wave grid
    const int m0 = blockIdx.y * 128, n0 = blockIdx.x * 128;

    v8f acc[4][2];
    const v8f zf = {0.f, 0.f, 0.f, 0.f, 0.f, 0.f, 0.f, 0.f};
#pragma unroll
    for (int mi = 0; mi < 4; ++mi)
#pragma unroll
        for (int ni = 0; ni < 2; ++ni) acc[mi][ni] = zf;

    const int fr = tid >> 3, fo = (tid & 7) * 8;       // 32 rows/pass, 8-elem chunks

    auto fill = [&](int buf, int ktile) {               // 8 async b128 per thread
#pragma unroll
        for (int it = 0; it < 4; ++it) {
            const int r = fr + it * 32;
            async_copy16(&sA[buf][r * GLDS + fo], A  + (size_t)(m0 + r) * K + ktile + fo);
            async_copy16(&sB[buf][r * GLDS + fo], Wt + (size_t)(n0 + r) * K + ktile + fo);
        }
    };

    fill(0, 0);                                         // prologue: tile 0
    int cur = 0;
    for (int kt = 0; kt < K; kt += 64) {
        __syncthreads();                                // prior reads of buf cur^1 done
        if (kt + 64 < K) {
            fill(cur ^ 1, kt + 64);                     // prefetch next tile
            async_wait8();                              // drain tile `cur` (in-order)
        } else {
            async_wait0();
        }
        __syncthreads();                                // tile `cur` visible to all waves

#pragma unroll
        for (int ks = 0; ks < 2; ++ks) {
            v16bf af[4];
#pragma unroll
            for (int mi = 0; mi < 4; ++mi) {
                const int row = wm * 64 + mi * 16 + ln;
                const unsigned short* p = &sA[cur][row * GLDS + ks * 32 + lh * 8];
                af[mi] = make_frag(p, p + 16);
            }
#pragma unroll
            for (int ni = 0; ni < 2; ++ni) {
                const int col = wn * 32 + ni * 16 + ln;
                const unsigned short* p = &sB[cur][col * GLDS + ks * 32 + lh * 8];
                v16bf bf = make_frag(p, p + 16);
#pragma unroll
                for (int mi = 0; mi < 4; ++mi)
                    acc[mi][ni] = __builtin_amdgcn_wmma_f32_16x16x32_bf16(
                        false, af[mi], false, bf, (short)0, acc[mi][ni], false, false);
            }
        }
        cur ^= 1;
    }

    // epilogue: bias, gelu(exact), residual, f32/bf16 store (all compile-time)
#pragma unroll
    for (int mi = 0; mi < 4; ++mi) {
#pragma unroll
        for (int ni = 0; ni < 2; ++ni) {
            const int gcol = n0 + wn * 32 + ni * 16 + ln;
            const float bv = BIAS ? bias[gcol] : 0.f;
#pragma unroll
            for (int j = 0; j < 8; ++j) {
                const int grow = m0 + wm * 64 + mi * 16 + lh * 8 + j;
                float v = acc[mi][ni][j] + bv;
                if (GELU) v = 0.5f * v * (1.f + erff(v * 0.70710678118654752f));
                if (RES)  v += residual[(size_t)grow * N + gcol];
                if (OUTF) Cf[(size_t)grow * N + gcol] = v;
                if (OUTB) Cbf[(size_t)grow * N + gcol] = f32_to_bf16(v);
            }
        }
    }
}

// ---------------------------------------------------------------------------
// Flash attention (WMMA): Q[B,NH,Lq,HD], K[B,NH,Lk,HD], Vt[B,NH,HD,Lk]
//   -> Out[B,Lq,NH*HD] (bf16). 128 q-rows/block, 8 waves x 16 q-rows.
// Double-buffered K/Vt tiles; causal mask analytic; optional key bias [B,Lk].
// ---------------------------------------------------------------------------
__global__ __launch_bounds__(256)
void flash_attn_kernel(const unsigned short* __restrict__ Q,
                       const unsigned short* __restrict__ Kg,
                       const unsigned short* __restrict__ VtG,
                       const float* __restrict__ key_bias,
                       unsigned short* __restrict__ Out,
                       int Lq, int Lk, int causal, float scale) {
    __shared__ __align__(16) unsigned short sQ[128 * GLDS];
    __shared__ __align__(16) unsigned short sK[2][64 * GLDS];
    __shared__ __align__(16) unsigned short sVt[2][64 * GLDS];   // [d][key]
    __shared__ __align__(16) unsigned short sP[128 * GLDS];

    const int tid = threadIdx.x;
    const int wave = tid >> 5, lane = tid & 31;
    const int lh = lane >> 4, ln = lane & 15;
    const int qt = blockIdx.x, h = blockIdx.y, b = blockIdx.z;
    const int q0 = qt * 128;

    const size_t qbase  = ((size_t)(b * NHEAD + h) * Lq + q0) * HDH;
    const size_t kbase  = (size_t)(b * NHEAD + h) * Lk * HDH;
    const size_t vtbase = (size_t)(b * NHEAD + h) * HDH * Lk;

    const int fr = tid >> 3, fo = (tid & 7) * 8;

    auto fillKV = [&](int buf, int k0) {                // 4 async b128 per thread
#pragma unroll
        for (int it = 0; it < 2; ++it) {
            const int r = fr + it * 32;
            async_copy16(&sK[buf][r * GLDS + fo],  Kg  + kbase  + (size_t)(k0 + r) * HDH + fo);
            async_copy16(&sVt[buf][r * GLDS + fo], VtG + vtbase + (size_t)r * Lk + k0 + fo);
        }
    };

    // prologue: Q tile (4 asyncs) + first K/V tile (4 asyncs)
#pragma unroll
    for (int it = 0; it < 4; ++it) {
        const int r = fr + it * 32;
        async_copy16(sQ + r * GLDS + fo, Q + qbase + (size_t)r * HDH + fo);
    }
    fillKV(0, 0);

    float m_s[8], l_s[8];
    v8f accO[4];
    const v8f zf = {0.f, 0.f, 0.f, 0.f, 0.f, 0.f, 0.f, 0.f};
#pragma unroll
    for (int j = 0; j < 8; ++j) { m_s[j] = -3.0e38f; l_s[j] = 0.f; }
#pragma unroll
    for (int nt = 0; nt < 4; ++nt) accO[nt] = zf;

    const int nkt = causal ? (q0 / 64 + 2) : (Lk / 64);
    int cur = 0;
    for (int kt = 0; kt < nkt; ++kt) {
        const int k0 = kt * 64;
        __syncthreads();                                // prior reads of buf cur^1 done
        if (kt + 1 < nkt) {
            fillKV(cur ^ 1, k0 + 64);                   // prefetch next K/V tile
            async_wait4();                              // drain Q + tile `cur` (in-order)
        } else {
            async_wait0();
        }
        __syncthreads();

        // S = Q K^T  (contraction over HD=64, two 32-wide WMMA steps)
        v8f sF[4];
#pragma unroll
        for (int kn = 0; kn < 4; ++kn) sF[kn] = zf;
        const int qrow = wave * 16 + ln;
#pragma unroll
        for (int ks = 0; ks < 2; ++ks) {
            const unsigned short* pa = sQ + qrow * GLDS + ks * 32 + lh * 8;
            v16bf a = make_frag(pa, pa + 16);
#pragma unroll
            for (int kn = 0; kn < 4; ++kn) {
                const int kcol = kn * 16 + ln;
                const unsigned short* pb = &sK[cur][kcol * GLDS + ks * 32 + lh * 8];
                v16bf bf = make_frag(pb, pb + 16);
                sF[kn] = __builtin_amdgcn_wmma_f32_16x16x32_bf16(
                    false, a, false, bf, (short)0, sF[kn], false, false);
            }
        }

        // online softmax update (rows live in 16-lane halves)
#pragma unroll
        for (int j = 0; j < 8; ++j) {
            const int qg = q0 + wave * 16 + lh * 8 + j;
            float pv[4];
            float rowm = -3.0e38f;
#pragma unroll
            for (int kn = 0; kn < 4; ++kn) {
                float v = sF[kn][j] * scale;
                const int kg = k0 + kn * 16 + ln;
                if (causal && kg > qg) v = -3.0e38f;
                if (key_bias) v += key_bias[(size_t)b * Lk + kg];
                pv[kn] = v;
                rowm = fmaxf(rowm, v);
            }
#pragma unroll
            for (int off = 8; off; off >>= 1)
                rowm = fmaxf(rowm, __shfl_xor(rowm, off, 32));
            const float mn = fmaxf(m_s[j], rowm);
            const float corr = __expf(m_s[j] - mn);
            float rs = 0.f;
#pragma unroll
            for (int kn = 0; kn < 4; ++kn) {
                const float p = __expf(pv[kn] - mn);
                rs += p;
                sP[(wave * 16 + lh * 8 + j) * GLDS + kn * 16 + ln] = f32_to_bf16(p);
            }
#pragma unroll
            for (int off = 8; off; off >>= 1) rs += __shfl_xor(rs, off, 32);
            l_s[j] = l_s[j] * corr + rs;
            m_s[j] = mn;
#pragma unroll
            for (int nt = 0; nt < 4; ++nt) accO[nt][j] *= corr;
        }

        // O += P @ V   (contraction over 64 keys; per-wave sP rows, DS in-order)
#pragma unroll
        for (int ks = 0; ks < 2; ++ks) {
            const unsigned short* pa = sP + (wave * 16 + ln) * GLDS + ks * 32 + lh * 8;
            v16bf a = make_frag(pa, pa + 16);
#pragma unroll
            for (int nt = 0; nt < 4; ++nt) {
                const int dcol = nt * 16 + ln;
                const unsigned short* pb = &sVt[cur][dcol * GLDS + ks * 32 + lh * 8];
                v16bf bf = make_frag(pb, pb + 16);
                accO[nt] = __builtin_amdgcn_wmma_f32_16x16x32_bf16(
                    false, a, false, bf, (short)0, accO[nt], false, false);
            }
        }
        cur ^= 1;
    }

    // write O / l  -> bf16 [B, Lq, NH*HD]
#pragma unroll
    for (int nt = 0; nt < 4; ++nt) {
#pragma unroll
        for (int j = 0; j < 8; ++j) {
            const int qg = q0 + wave * 16 + lh * 8 + j;
            const float o = accO[nt][j] / l_s[j];
            Out[((size_t)b * Lq + qg) * (NHEAD * HDH) + h * HDH + nt * 16 + ln] =
                f32_to_bf16(o);
        }
    }
}

// ---------------------------------------------------------------------------
// TDM probe (never launched): validates __builtin_amdgcn_tensor_load_to_lds
// (6-arg amdgpu-toolchain form) with an honest 2D-tile descriptor.
// ---------------------------------------------------------------------------
__global__ void tdm_probe_kernel(const unsigned short* __restrict__ src,
                                 unsigned short* __restrict__ dst,
                                 int tensorK, int tensorM) {
    __shared__ __align__(16) unsigned short sT[64 * 64];
    unsigned long long ga = (unsigned long long)src;
    unsigned ldsa = (unsigned)(unsigned long long)(&sT[0]);
    v4u g0 = { 1u,                                  // count=1, user mode
               ldsa,                                // lds_addr
               (unsigned)ga,                        // global_addr[31:0]
               (unsigned)((ga >> 32) & 0x1FFFFFFu) | (2u << 30) };  // addr[56:32] | type=2
    v8i g1;
    g1[0] = (int)(1u << 16);                        // data_size=2B; no pad/iterate
    g1[1] = (int)(((unsigned)tensorK & 0xFFFFu) << 16);          // tensor_dim0[15:0]
    g1[2] = (int)((((unsigned)tensorK >> 16) & 0xFFFFu)
                  | (((unsigned)tensorM & 0xFFFFu) << 16));      // td0[31:16] | td1[15:0]
    g1[3] = (int)((((unsigned)tensorM >> 16) & 0xFFFFu)
                  | (64u << 16));                                // td1[31:16] | tile_dim0=64
    g1[4] = (int)(64u);                                          // tile_dim1=64, tile_dim2=0
    g1[5] = (int)((unsigned)tensorK);                            // dim0_stride[31:0]
    g1[6] = 0;                                                   // stride[47:32] | d1stride lo
    g1[7] = 0;
    v4i gz = {0, 0, 0, 0};
    v8i gz8 = {0, 0, 0, 0, 0, 0, 0, 0};
    __builtin_amdgcn_tensor_load_to_lds(g0, g1, gz, gz, gz8, 0);
    __builtin_amdgcn_s_wait_tensorcnt((short)0);
    __syncthreads();
    for (int i = threadIdx.x; i < 64 * 64; i += blockDim.x) dst[i] = sT[i];
}

// ---------------------------------------------------------------------------
// Host-side orchestration
// ---------------------------------------------------------------------------
static const size_t MBs   = 1024ull * 1024ull;
static const size_t O_WQKV = 0;                                    // 6 MB
static const size_t O_WOUT = O_WQKV + (size_t)Hdim * 3 * Hdim * 2; // 2 MB
static const size_t O_WCQ  = O_WOUT + (size_t)Hdim * Hdim * 2;
static const size_t O_WCKV = O_WCQ  + (size_t)Hdim * Hdim * 2;     // 4 MB
static const size_t O_WCOUT= O_WCKV + (size_t)Hdim * 2 * Hdim * 2;
static const size_t O_WFF1 = O_WCOUT+ (size_t)Hdim * Hdim * 2;     // 8 MB
static const size_t O_WFF2 = O_WFF1 + (size_t)Hdim * 4 * Hdim * 2; // 8 MB
static const size_t O_SCR  = O_WFF2 + (size_t)4 * Hdim * Hdim * 2; // = 32 MB
static const size_t O_ACT  = O_SCR + 48 * MBs;   // ln outputs (bf16, 8 MB)
static const size_t O_Q    = O_ACT + 8 * MBs;    // [B,NH,S,HD] bf16
static const size_t O_K    = O_Q   + 8 * MBs;
static const size_t O_V    = O_K   + 8 * MBs;    // d-major [B,NH,HD,S]
static const size_t O_ATT  = O_V   + 8 * MBs;    // attn out bf16
static const size_t O_X1   = O_ATT + 8 * MBs;    // f32 16 MB
static const size_t O_X2   = O_X1  + 16 * MBs;   // f32 16 MB  (total 152 MB)

extern "C" void kernel_launch(void* const* d_in, const int* in_sizes, int n_in,
                              void* d_out, int out_size, void* d_ws, size_t ws_size,
                              hipStream_t stream) {
    const float* x     = (const float*)d_in[0];
    const float* rope  = (const float*)d_in[1];
    // d_in[2] attention_mask: causal, applied analytically
    const float* cond  = (const float*)d_in[3];
    const float* cmask = (const float*)d_in[4];
    const float* ln1g = (const float*)d_in[5],  *ln1b = (const float*)d_in[6];
    const float* ln2g = (const float*)d_in[7],  *ln2b = (const float*)d_in[8];
    const float* lncg = (const float*)d_in[9],  *lncb = (const float*)d_in[10];
    const float* w_qkv = (const float*)d_in[11];
    const float* w_out = (const float*)d_in[12], *b_out = (const float*)d_in[13];
    const float* w_cq  = (const float*)d_in[14];
    const float* w_ckv = (const float*)d_in[15];
    const float* w_cout= (const float*)d_in[16], *b_cout= (const float*)d_in[17];
    const float* w_ff1 = (const float*)d_in[18], *b_ff1 = (const float*)d_in[19];
    const float* w_ff2 = (const float*)d_in[20], *b_ff2 = (const float*)d_in[21];

    char* ws = (char*)d_ws;
    unsigned short* WQKV = (unsigned short*)(ws + O_WQKV);   // all [N][K] bf16
    unsigned short* WOUT = (unsigned short*)(ws + O_WOUT);
    unsigned short* WCQ  = (unsigned short*)(ws + O_WCQ);
    unsigned short* WCKV = (unsigned short*)(ws + O_WCKV);
    unsigned short* WCOUT= (unsigned short*)(ws + O_WCOUT);
    unsigned short* WFF1 = (unsigned short*)(ws + O_WFF1);
    unsigned short* WFF2 = (unsigned short*)(ws + O_WFF2);
    float*          QKVF = (float*)(ws + O_SCR);                  // [4096,3072]
    float*          QCF  = (float*)(ws + O_SCR);                  // reuse
    float*          KVF  = (float*)(ws + O_SCR);                  // reuse
    unsigned short* CONDB= (unsigned short*)(ws + O_SCR + 40 * MBs);
    unsigned short* HFF  = (unsigned short*)(ws + O_SCR);         // reuse [4096,4096]
    unsigned short* ACT  = (unsigned short*)(ws + O_ACT);
    unsigned short* Qh   = (unsigned short*)(ws + O_Q);
    unsigned short* Kh   = (unsigned short*)(ws + O_K);
    unsigned short* Vth  = (unsigned short*)(ws + O_V);
    unsigned short* ATT  = (unsigned short*)(ws + O_ATT);
    float*          X1   = (float*)(ws + O_X1);
    float*          X2   = (float*)(ws + O_X2);

    const int M = Bb * Ss;  // 4096
    const float scale = 0.125f;  // 1/sqrt(64)

    // --- weight transpose+convert to bf16 [N][K] (fits L2) ---
    transpose_convert_kernel<<<dim3(3 * Hdim / 32, Hdim / 32), 256, 0, stream>>>(w_qkv, WQKV, Hdim, 3 * Hdim);
    transpose_convert_kernel<<<dim3(Hdim / 32, Hdim / 32), 256, 0, stream>>>(w_out, WOUT, Hdim, Hdim);
    transpose_convert_kernel<<<dim3(Hdim / 32, Hdim / 32), 256, 0, stream>>>(w_cq,  WCQ,  Hdim, Hdim);
    transpose_convert_kernel<<<dim3(2 * Hdim / 32, Hdim / 32), 256, 0, stream>>>(w_ckv, WCKV, Hdim, 2 * Hdim);
    transpose_convert_kernel<<<dim3(Hdim / 32, Hdim / 32), 256, 0, stream>>>(w_cout, WCOUT, Hdim, Hdim);
    transpose_convert_kernel<<<dim3(4 * Hdim / 32, Hdim / 32), 256, 0, stream>>>(w_ff1, WFF1, Hdim, 4 * Hdim);
    transpose_convert_kernel<<<dim3(Hdim / 32, 4 * Hdim / 32), 256, 0, stream>>>(w_ff2, WFF2, 4 * Hdim, Hdim);

    // --- self attention ---
    layernorm_kernel<<<M, 256, 0, stream>>>(x, ln1g, ln1b, ACT);
    gemm_bf16_kernel<0,0,0,1,0><<<dim3(3 * Hdim / 128, M / 128), 256, 0, stream>>>(
        ACT, WQKV, nullptr, nullptr, QKVF, nullptr, M, 3 * Hdim, Hdim);
    rope_split_kernel<<<dim3(Ss, Bb), 256, 0, stream>>>(QKVF, rope, Qh, Kh, Vth);
    flash_attn_kernel<<<dim3(Ss / 128, NHEAD, Bb), 256, 0, stream>>>(
        Qh, Kh, Vth, nullptr, ATT, Ss, Ss, 1, scale);
    gemm_bf16_kernel<1,1,0,1,0><<<dim3(Hdim / 128, M / 128), 256, 0, stream>>>(
        ATT, WOUT, b_out, x, X1, nullptr, M, Hdim, Hdim);

    // --- cross attention ---
    layernorm_kernel<<<M, 256, 0, stream>>>(X1, lncg, lncb, ACT);
    gemm_bf16_kernel<0,0,0,1,0><<<dim3(Hdim / 128, M / 128), 256, 0, stream>>>(
        ACT, WCQ, nullptr, nullptr, QCF, nullptr, M, Hdim, Hdim);
    reshape_heads_kernel<<<dim3(Ss, Bb), 256, 0, stream>>>(QCF, Qh, Ss, 1, 0, 0);
    convert_bf16_kernel<<<512, 256, 0, stream>>>(cond, CONDB, (size_t)Bb * Cc * Hdim);
    gemm_bf16_kernel<0,0,0,1,0><<<dim3(2 * Hdim / 128, (Bb * Cc) / 128), 256, 0, stream>>>(
        CONDB, WCKV, nullptr, nullptr, KVF, nullptr, Bb * Cc, 2 * Hdim, Hdim);
    reshape_heads_kernel<<<dim3(Cc, Bb), 256, 0, stream>>>(KVF, Kh, Cc, 2, 0, 0);
    reshape_heads_kernel<<<dim3(Cc, Bb), 256, 0, stream>>>(KVF, Vth, Cc, 2, 1, 1);
    flash_attn_kernel<<<dim3(Ss / 128, NHEAD, Bb), 256, 0, stream>>>(
        Qh, Kh, Vth, cmask, ATT, Ss, Cc, 0, scale);
    gemm_bf16_kernel<1,1,0,1,0><<<dim3(Hdim / 128, M / 128), 256, 0, stream>>>(
        ATT, WCOUT, b_cout, X1, X2, nullptr, M, Hdim, Hdim);

    // --- FFN ---
    layernorm_kernel<<<M, 256, 0, stream>>>(X2, ln2g, ln2b, ACT);
    gemm_bf16_kernel<1,0,1,0,1><<<dim3(4 * Hdim / 128, M / 128), 256, 0, stream>>>(
        ACT, WFF1, b_ff1, nullptr, nullptr, HFF, M, 4 * Hdim, Hdim);
    gemm_bf16_kernel<1,1,0,1,0><<<dim3(Hdim / 128, M / 128), 256, 0, stream>>>(
        HFF, WFF2, b_ff2, X2, (float*)d_out, nullptr, M, Hdim, 4 * Hdim);
}
